// BitLinear_40389872452002
// MI455X (gfx1250) — matrix operand order repaired and checked by reading, not compiled, not run on hardware
//
#include <hip/hip_runtime.h>

// BitLinear: out[16384, 4096] = x[16384, 4096] @ W[4096, 4096] + bias
// W is ternary {-1,0,1} -> exact in f16.  WMMA f32_16x16x32_f16 path.

typedef __attribute__((ext_vector_type(16))) _Float16 v16h;
typedef __attribute__((ext_vector_type(8)))  _Float16 v8h;
typedef __attribute__((ext_vector_type(4)))  _Float16 v4h;
typedef __attribute__((ext_vector_type(8)))  float    v8f;
typedef __attribute__((ext_vector_type(4)))  float    v4f;

#define M_DIM 16384
#define N_DIM 4096
#define K_DIM 4096

#define BM 128
#define BN 128
#define BK 64
#define LDS_K (BK + 8)     // 72 halves = 144 bytes per row: 16B-aligned, conflict-breaking pad
#define NTHREADS 256       // 8 wave32 waves

__global__ __launch_bounds__(NTHREADS)
void bitlinear_wmma_f16(const float* __restrict__ X,
                        const float* __restrict__ W,
                        const float* __restrict__ bias,
                        float* __restrict__ out)
{
    __shared__ __align__(16) _Float16 sA[BM * LDS_K];  // [m][k]
    __shared__ __align__(16) _Float16 sB[BN * LDS_K];  // [n][k] (transposed W tile)

    const int t    = threadIdx.x;
    const int lane = t & 31;
    const int wave = t >> 5;
    const int ln   = lane & 15;   // N (or M) index within 16x16 tile
    const int lg   = lane >> 4;   // lane group 0/1

    // 8 waves: 2 along M (64 rows each) x 4 along N (32 cols each)
    const int waveM = (wave >> 2) * 64;
    const int waveN = (wave & 3) * 32;

    const int blockN = blockIdx.x * BN;
    const int blockM = blockIdx.y * BM;

    // staging decomposition
    const int a_kg = t & 15;   // 16 groups of 4 consecutive k  (64 k total)
    const int a_m0 = t >> 4;   // row 0..15, step 16
    const int b_ng = t & 31;   // 32 groups of 4 consecutive n  (128 n total)
    const int b_k0 = t >> 5;   // k row 0..7, step 8

    v8f acc[4][2] = {};        // 4 m-tiles x 2 n-tiles of 16x16 f32

    const float* Xt = X + (size_t)blockM * K_DIM;
    const float* Wt = W + blockN;

    for (int k0 = 0; k0 < K_DIM; k0 += BK) {
        __syncthreads();

        // ---- stage A: 128x64 fp32 -> f16 into sA[m][k] ----
        #pragma unroll
        for (int mm = 0; mm < BM; mm += 16) {
            const int m = a_m0 + mm;
            v4f f = *(const v4f*)(Xt + (size_t)m * K_DIM + (k0 + a_kg * 4));
            v4h h = { (_Float16)f.x, (_Float16)f.y, (_Float16)f.z, (_Float16)f.w };
            *(v4h*)(&sA[m * LDS_K + a_kg * 4]) = h;   // ds_store_b64
        }

        // ---- stage B: 64x128 fp32 -> f16, transposed into sB[n][k] ----
        #pragma unroll
        for (int kk = 0; kk < BK; kk += 8) {
            const int k = b_k0 + kk;
            v4f f = *(const v4f*)(Wt + (size_t)(k0 + k) * N_DIM + b_ng * 4);
            sB[(b_ng * 4 + 0) * LDS_K + k] = (_Float16)f.x;
            sB[(b_ng * 4 + 1) * LDS_K + k] = (_Float16)f.y;
            sB[(b_ng * 4 + 2) * LDS_K + k] = (_Float16)f.z;
            sB[(b_ng * 4 + 3) * LDS_K + k] = (_Float16)f.w;
        }

        // prefetch next K-slab while we compute (global_prefetch_b8)
        if (k0 + BK < K_DIM) {
            __builtin_prefetch(Xt + (size_t)a_m0 * K_DIM + (k0 + BK + a_kg * 4), 0, 1);
            __builtin_prefetch(Wt + (size_t)(k0 + BK + b_k0) * N_DIM + b_ng * 4, 0, 1);
        }

        __syncthreads();

        // ---- compute: 2 K-slices of 32, 8 WMMAs each ----
        #pragma unroll
        for (int ko = 0; ko < BK; ko += 32) {
            v16h a[4];
            v16h b[2];

            // A fragment, 16x32 f16: lanes 0-15 hold K 0-7 & 16-23, lanes 16-31 hold K 8-15 & 24-31
            #pragma unroll
            for (int i = 0; i < 4; ++i) {
                const _Float16* pa = &sA[(waveM + i * 16 + ln) * LDS_K + ko + lg * 8];
                v8h lo = *(const v8h*)pa;          // ds_load_b128
                v8h hi = *(const v8h*)(pa + 16);   // ds_load_b128
                a[i] = __builtin_shufflevector(lo, hi, 0,1,2,3,4,5,6,7,8,9,10,11,12,13,14,15);
            }
            // B fragment, 32x16 f16: lane = column N, lanes 0-15 hold K 0-7 & 16-23, lanes 16-31 hold K 8-15 & 24-31
            #pragma unroll
            for (int j = 0; j < 2; ++j) {
                const _Float16* pb = &sB[(waveN + j * 16 + ln) * LDS_K + ko + lg * 8];
                v8h lo = *(const v8h*)pb;          // ds_load_b128
                v8h hi = *(const v8h*)(pb + 16);   // ds_load_b128
                b[j] = __builtin_shufflevector(lo, hi, 0,1,2,3,4,5,6,7,8,9,10,11,12,13,14,15);
            }

            #pragma unroll
            for (int i = 0; i < 4; ++i)
                #pragma unroll
                for (int j = 0; j < 2; ++j)
                    acc[i][j] = __builtin_amdgcn_wmma_f32_16x16x32_f16(
                        /*neg_a=*/false, a[i], /*neg_b=*/false, b[j],
                        /*c_mod=*/(short)0, acc[i][j],
                        /*reuse_a=*/false, /*reuse_b=*/false);
        }
    }

    // ---- epilogue: C/D layout: VGPR r -> M = r + 8*lanegroup, N = lane&15 ----
    #pragma unroll
    for (int j = 0; j < 2; ++j) {
        const int n = blockN + waveN + j * 16 + ln;
        const float bv = bias[n];
        #pragma unroll
        for (int i = 0; i < 4; ++i) {
            const size_t mbase = (size_t)(blockM + waveM + i * 16 + lg * 8);
            float* op = out + mbase * N_DIM + n;
            #pragma unroll
            for (int r = 0; r < 8; ++r)
                op[(size_t)r * N_DIM] = acc[i][j][r] + bv;
        }
    }
}

extern "C" void kernel_launch(void* const* d_in, const int* in_sizes, int n_in,
                              void* d_out, int out_size, void* d_ws, size_t ws_size,
                              hipStream_t stream) {
    (void)in_sizes; (void)n_in; (void)d_ws; (void)ws_size; (void)out_size;
    const float* x    = (const float*)d_in[0];
    const float* w    = (const float*)d_in[1];
    const float* bias = (const float*)d_in[2];
    float* out        = (float*)d_out;

    dim3 grid(N_DIM / BN, M_DIM / BM);   // x-major blocks share the A tile -> x rows stay in L2
    dim3 block(NTHREADS);
    bitlinear_wmma_f16<<<grid, block, 0, stream>>>(x, w, bias, out);
}
